// ExpertFeedForward_65618510348722
// MI455X (gfx1250) — matrix operand order repaired and checked by reading, not compiled, not run on hardware
//
#include <hip/hip_runtime.h>
#include <cstdint>
#include <cstddef>

typedef __attribute__((ext_vector_type(16))) __bf16 v16bf;
typedef __attribute__((ext_vector_type(8)))  __bf16 v8bf;
typedef __attribute__((ext_vector_type(8)))  float  v8f;

static constexpr int B_ = 2, S_ = 2048, D_ = 1024, F_ = 4096, E_ = 8;
static constexpr int NTOK = B_ * S_;        // 4096 tokens
static constexpr int KT1 = D_ / 32;         // 32  K-tiles for GEMM1
static constexpr int NT1 = F_ / 16;         // 256 N-tiles for GEMM1
static constexpr int KT2 = F_ / 32;         // 128 K-tiles for GEMM2
static constexpr int NT2 = D_ / 16;         // 64  N-tiles for GEMM2
static constexpr int TM  = 32;              // tokens per block (2 row-tiles)

// -------- workspace layout (bytes) --------
static constexpr size_t WS_CNT  = 0;                                  // 8 ints
static constexpr size_t WS_TOK  = 256;                                // E*NTOK ints
static constexpr size_t WS_PROB = WS_TOK + (size_t)E_ * NTOK * 4;     // E*NTOK floats
static constexpr size_t WS_XB   = (size_t)1 << 19;                    // NTOK*D bf16 (8MB)
static constexpr size_t WS_W1P  = (size_t)16 << 20;                   // 64MB bf16, permuted
static constexpr size_t WS_W2P  = (size_t)80 << 20;                   // 64MB bf16, permuted

// dynamic LDS: xa (2 row-tiles, A-frag layout) + ha (h chunk, A-frag layout)
static constexpr int XA_ELEMS = 2 * KT1 * 32 * 16;   // 32768  (64KB)
static constexpr int HA_ELEMS = 2 * 4 * 32 * 16;     // 4096   (8KB)
static constexpr size_t SMEM_BYTES = (size_t)(XA_ELEMS + HA_ELEMS) * sizeof(__bf16);

// ---------------- zero out + counters ----------------
__global__ void zero_kernel(float* out, int n, int* cnt) {
    int stride = gridDim.x * blockDim.x;
    for (int i = blockIdx.x * blockDim.x + threadIdx.x; i < n; i += stride)
        out[i] = 0.0f;
    if (blockIdx.x == 0 && threadIdx.x < E_) cnt[threadIdx.x] = 0;
}

// ---------------- fp32 -> bf16 for x ----------------
__global__ void convert_x(const float* __restrict__ x, __bf16* __restrict__ xb, int n) {
    int stride = gridDim.x * blockDim.x;
    for (int i = blockIdx.x * blockDim.x + threadIdx.x; i < n; i += stride)
        xb[i] = (__bf16)x[i];
}

// ------ W1 [E][D][F] -> bf16 B-fragment layout [E][KT1][NT1][lane(32)][el(16)] ------
__global__ void permute_w1(const float* __restrict__ W1, __bf16* __restrict__ W1p) {
    const int total = E_ * D_ * F_;                     // 33,554,432
    int stride = gridDim.x * blockDim.x;
    for (int i = blockIdx.x * blockDim.x + threadIdx.x; i < total; i += stride) {
        int e   = i >> 22;                              // per-expert = 2^22 elems
        int r   = i & ((1 << 22) - 1);
        int kt  = r >> 17;                              // NT1*512 = 2^17
        int r2  = r & ((1 << 17) - 1);
        int nt  = r2 >> 9;
        int ln  = (r2 >> 4) & 31;
        int el  = r2 & 15;
        int k   = kt * 32 + ((el & 8) << 1) + ((ln & 16) >> 1) + (el & 7); // B layout: K = 16*(el>=8) + 8*half + (el&7)
        int n   = nt * 16 + (ln & 15);
        W1p[i]  = (__bf16)W1[((size_t)e * D_ + k) * F_ + n];
    }
}

// ------ W2 [E][F][D] -> bf16 B-fragment layout [E][KT2][NT2][lane][el] ------
__global__ void permute_w2(const float* __restrict__ W2, __bf16* __restrict__ W2p) {
    const int total = E_ * F_ * D_;
    int stride = gridDim.x * blockDim.x;
    for (int i = blockIdx.x * blockDim.x + threadIdx.x; i < total; i += stride) {
        int e   = i >> 22;
        int r   = i & ((1 << 22) - 1);
        int kt  = r >> 15;                              // NT2*512 = 2^15
        int r2  = r & ((1 << 15) - 1);
        int nt  = r2 >> 9;
        int ln  = (r2 >> 4) & 31;
        int el  = r2 & 15;
        int k   = kt * 32 + ((el & 8) << 1) + ((ln & 16) >> 1) + (el & 7); // K in F dim
        int n   = nt * 16 + (ln & 15);                  // N in D dim
        W2p[i]  = (__bf16)W2[((size_t)e * F_ + k) * D_ + n];
    }
}

// ---------------- gate + top-2 + softmax + compaction ----------------
__global__ void gate_topk(const float* __restrict__ x, const float* __restrict__ Wg,
                          const float* __restrict__ bg, int* __restrict__ cnt,
                          int* __restrict__ tokList, float* __restrict__ probList) {
    int tok = blockIdx.x * blockDim.x + threadIdx.x;
    if (tok >= NTOK) return;
    const float* xr = x + (size_t)tok * D_;
    float acc[E_];
    #pragma unroll
    for (int e = 0; e < E_; ++e) acc[e] = bg[e];
    for (int d = 0; d < D_; ++d) {
        float xv = xr[d];
        #pragma unroll
        for (int e = 0; e < E_; ++e) acc[e] += xv * Wg[d * E_ + e];
    }
    int i0 = 0; float s0 = acc[0];
    #pragma unroll
    for (int e = 1; e < E_; ++e) if (acc[e] > s0) { s0 = acc[e]; i0 = e; }
    int i1 = -1; float s1 = -3.4e38f;
    #pragma unroll
    for (int e = 0; e < E_; ++e) if (e != i0 && acc[e] > s1) { s1 = acc[e]; i1 = e; }
    float t  = __expf(s1 - s0);        // s0 >= s1
    float p0 = 1.0f / (1.0f + t);
    float p1 = t * p0;
    int slot0 = atomicAdd(&cnt[i0], 1);
    tokList[i0 * NTOK + slot0]  = tok;
    probList[i0 * NTOK + slot0] = p0;
    int slot1 = atomicAdd(&cnt[i1], 1);
    tokList[i1 * NTOK + slot1]  = tok;
    probList[i1 * NTOK + slot1] = p1;
}

// ---------------- routed expert FFN, bf16 WMMA, M=32 ----------------
// block = 256 threads (8 waves); grid = (NTOK/TM tiles, E experts)
__global__ void __launch_bounds__(256)
moe_ffn(const __bf16* __restrict__ xb,
        const __bf16* __restrict__ W1p,
        const __bf16* __restrict__ W2p,
        const float* __restrict__ b1, const float* __restrict__ b2,
        const int* __restrict__ cnt, const int* __restrict__ tokList,
        const float* __restrict__ probList, float* __restrict__ out) {
    const int e     = blockIdx.y;
    const int tile  = blockIdx.x;
    const int count = cnt[e];
    if (tile * TM >= count) return;

    extern __shared__ __align__(32) char smem_raw[];
    __bf16* xa = (__bf16*)smem_raw;              // [tr][kt][lane][el]  64KB
    __bf16* ha = xa + XA_ELEMS;                  // [tr][ktf][lane][el]  8KB
    __shared__ int   tokS[TM];
    __shared__ float pS[TM];

    const int t = threadIdx.x;
    if (t < TM) {
        int idx = tile * TM + t;
        bool v = idx < count;
        tokS[t] = v ? tokList[e * NTOK + idx] : 0;
        pS[t]   = v ? probList[e * NTOK + idx] : 0.0f;
    }
    __syncthreads();

    // stage x tile into LDS in A-fragment layout, 8-element (16B) runs:
    // A layout: k = kt*32 + 2*(el&8) + 8*(lane>=16) + (el&7)  -> contiguous per el-octet
    for (int c = t; c < XA_ELEMS / 8; c += 256) {        // 4096 chunks, 16 per thread
        int oct = c & 1;                 // el 0-7 vs 8-15
        int ln  = (c >> 1) & 31;
        int kt  = (c >> 6) & 31;
        int tr  = c >> 11;
        int row = tr * 16 + (ln & 15);
        int k0  = kt * 32 + (oct << 4) + ((ln & 16) >> 1);
        const v8bf* src = (const v8bf*)&xb[(size_t)tokS[row] * D_ + k0];
        v8bf* dst = (v8bf*)&xa[(size_t)c * 8];
        *dst = *src;
    }
    __syncthreads();

    const int w    = t >> 5;        // wave id 0..7
    const int lane = t & 31;
    const int col  = lane & 15;
    const int hs   = lane >> 4;     // 0 or 1 (row half)

    v8f yacc[2][8];
    #pragma unroll
    for (int tr = 0; tr < 2; ++tr)
        #pragma unroll
        for (int i = 0; i < 8; ++i) yacc[tr][i] = {};

    for (int fcChunk = 0; fcChunk < F_ / 128; ++fcChunk) {
        // ---- GEMM1: h[32 x 16] for this wave's column tile, K = D = 1024 ----
        v8f c0 = {}, c1 = {};
        const int nt1 = fcChunk * 8 + w;
        const __bf16* w1col = W1p + ((size_t)e * KT1 * NT1 + nt1) * 512 + (size_t)lane * 16;
        #pragma unroll 4
        for (int kt = 0; kt < KT1; ++kt) {
            v16bf b  = *(const v16bf*)(w1col + (size_t)kt * NT1 * 512);
            v16bf a0 = *(const v16bf*)&xa[(kt * 32 + lane) * 16];
            v16bf a1 = *(const v16bf*)&xa[((KT1 + kt) * 32 + lane) * 16];
            c0 = __builtin_amdgcn_wmma_f32_16x16x32_bf16(false, a0, false, b,
                                                         (short)0, c0, false, false);
            c1 = __builtin_amdgcn_wmma_f32_16x16x32_bf16(false, a1, false, b,
                                                         (short)0, c1, false, false);
        }
        float bias1 = b1[e * F_ + nt1 * 16 + col];
        // relu + bf16 + scatter into A-fragment layout for GEMM2
        const int k_chunk = w * 16 + col;       // K position within 128-chunk
        const int ktf     = k_chunk >> 5;
        const int k_loc   = k_chunk & 31;
        const int lane_d  = ((k_loc & 8) << 1); // +row below
        const int el2     = ((k_loc & 16) >> 1) | (k_loc & 7);
        #pragma unroll
        for (int r = 0; r < 8; ++r) {
            int rr = r + 8 * hs;
            float v0 = fmaxf(c0[r] + bias1, 0.0f);
            float v1 = fmaxf(c1[r] + bias1, 0.0f);
            ha[(ktf * 32 + lane_d + rr) * 16 + el2]       = (__bf16)v0;
            ha[((4 + ktf) * 32 + lane_d + rr) * 16 + el2] = (__bf16)v1;
        }
        __syncthreads();

        // ---- GEMM2 partial: y[32 x 128 (this wave)] += h_chunk @ W2chunk ----
        // ktf outer / nt inner: each A-fragment pair loaded from LDS once,
        // weight B-fragments streamed in the inner loop.
        const int ktG0 = fcChunk * 4;
        #pragma unroll
        for (int ktf2 = 0; ktf2 < 4; ++ktf2) {
            v16bf a20 = *(const v16bf*)&ha[(ktf2 * 32 + lane) * 16];
            v16bf a21 = *(const v16bf*)&ha[((4 + ktf2) * 32 + lane) * 16];
            const __bf16* w2row = W2p + (((size_t)e * KT2 + ktG0 + ktf2) * NT2 + w * 8) * 512
                                      + (size_t)lane * 16;
            #pragma unroll
            for (int nt = 0; nt < 8; ++nt) {
                v16bf b2f = *(const v16bf*)(w2row + (size_t)nt * 512);
                yacc[0][nt] = __builtin_amdgcn_wmma_f32_16x16x32_bf16(false, a20, false, b2f,
                                                                      (short)0, yacc[0][nt], false, false);
                yacc[1][nt] = __builtin_amdgcn_wmma_f32_16x16x32_bf16(false, a21, false, b2f,
                                                                      (short)0, yacc[1][nt], false, false);
            }
        }
        __syncthreads();   // before ha is overwritten next chunk
    }

    // ---- epilogue: + b2, scale by routed prob, atomic accumulate ----
    #pragma unroll
    for (int tr = 0; tr < 2; ++tr) {
        #pragma unroll
        for (int nt = 0; nt < 8; ++nt) {
            int n = w * 128 + nt * 16 + col;
            float bias2 = b2[e * D_ + n];
            #pragma unroll
            for (int r = 0; r < 8; ++r) {
                int row = tr * 16 + r + 8 * hs;
                float val = (yacc[tr][nt][r] + bias2) * pS[row];
                unsafeAtomicAdd(&out[(size_t)tokS[row] * D_ + n], val);
            }
        }
    }
}

extern "C" void kernel_launch(void* const* d_in, const int* in_sizes, int n_in,
                              void* d_out, int out_size, void* d_ws, size_t ws_size,
                              hipStream_t stream) {
    const float* x  = (const float*)d_in[0];
    const float* W1 = (const float*)d_in[1];
    const float* b1 = (const float*)d_in[2];
    const float* W2 = (const float*)d_in[3];
    const float* b2 = (const float*)d_in[4];
    const float* Wg = (const float*)d_in[5];
    const float* bg = (const float*)d_in[6];
    float* out = (float*)d_out;

    char* ws = (char*)d_ws;
    int*    cnt      = (int*)(ws + WS_CNT);
    int*    tokList  = (int*)(ws + WS_TOK);
    float*  probList = (float*)(ws + WS_PROB);
    __bf16* xb       = (__bf16*)(ws + WS_XB);
    __bf16* W1p      = (__bf16*)(ws + WS_W1P);
    __bf16* W2p      = (__bf16*)(ws + WS_W2P);

    zero_kernel<<<2048, 256, 0, stream>>>(out, out_size, cnt);
    convert_x  <<<2048, 256, 0, stream>>>(x, xb, NTOK * D_);
    permute_w1 <<<8192, 256, 0, stream>>>(W1, W1p);
    permute_w2 <<<8192, 256, 0, stream>>>(W2, W2p);
    gate_topk  <<<NTOK / 256, 256, 0, stream>>>(x, Wg, bg, cnt, tokList, probList);
    dim3 grid(NTOK / TM, E_);
    moe_ffn    <<<grid, 256, SMEM_BYTES, stream>>>(xb, W1p, W2p, b1, b2, cnt,
                                                   tokList, probList, out);
}
